// Model_70308614635674
// MI455X (gfx1250) — compile-verified, matrix operand order checked by reading, not run on hardware
//
#include <hip/hip_runtime.h>
#include <math.h>

// ---------------- problem constants (fixed by the reference) ----------------
#define BATCH    2048
#define UDIM     128
#define TITLE_L  10
#define WEMB     50
#define NB       30            // neighbors (both sides)
#define DIM      128           // NCAPS*NHID
#define NCAPS    8
#define NHID     16
#define FLATK    992           // 8 * 4 * 31
#define ROWS     (BATCH + BATCH*NB)   // 63488 encoder rows per side
#define ROUTIT   4

typedef __attribute__((ext_vector_type(16))) _Float16 v16h;
typedef __attribute__((ext_vector_type(8)))  _Float16 v8h;
typedef __attribute__((ext_vector_type(8)))  float    v8f;

// ---------------- WMMA helpers (gfx1250, wave32) ----------------
__device__ __forceinline__ v8f wmma16(v16h a, v16h b, v8f c) {
  // D = A(16x32 f16) * B(32x16 f16) + C(16x16 f32)
  return __builtin_amdgcn_wmma_f32_16x16x32_f16(false, a, false, b, (short)0, c,
                                                false, false);
}

__device__ __forceinline__ v8f bcast8(float v) {
  v8f r; for (int i = 0; i < 8; ++i) r[i] = v; return r;
}

// A fragment from row-major fp16 [16][stride] (LDS). Lane holds M = lane&15,
// K halves {base8..base8+7} and {base8+16..base8+23} per the ISA A layout.
__device__ __forceinline__ v16h load_a_frag(const _Float16* a, int laneM,
                                            int base8, int kt, int stride) {
  const _Float16* p = a + laneM * stride + kt * 32 + base8;
  v8h lo = *(const v8h*)p;
  v8h hi = *(const v8h*)(p + 16);
  return __builtin_shufflevector(lo, hi, 0,1,2,3,4,5,6,7,8,9,10,11,12,13,14,15);
}

// ---------------- K0: pack f32 [K][N] weights -> fp16 WMMA B fragments ------
// out[((kt*NT+nt)*32 + lane)*16 + h] = W[kt*32 + K(lane,h)][nt*16 + (lane&15)]
__global__ void swizzle_w_kernel(const float* __restrict__ W,
                                 _Float16* __restrict__ out,
                                 int NT, int N, int total) {
  int id = blockIdx.x * blockDim.x + threadIdx.x;
  if (id >= total) return;
  int h    = id & 15;
  int l    = (id >> 4) & 31;
  int frag = id >> 9;
  int nt = frag % NT;
  int kt = frag / NT;
  int n    = nt * 16 + (l & 15);
  int base = (l < 16) ? 0 : 8;
  int k    = kt * 32 + base + h + ((h >= 8) ? 8 : 0);
  out[id] = (_Float16)W[k * N + n];
}

// ---------------- K1: fused title-CNN + dense(992->128) + item(128->128) ----
__global__ __launch_bounds__(256) void news_encoder_kernel(
    const int* __restrict__ news_indices, const int* __restrict__ user_indices,
    const int* __restrict__ user_news,    const int* __restrict__ title,
    const float* __restrict__ word_emb,
    const float* __restrict__ conv_w,     const float* __restrict__ conv_b,
    const _Float16* __restrict__ dsw,     const float* __restrict__ dense_b,
    const _Float16* __restrict__ isw,     const float* __restrict__ item_b,
    _Float16* __restrict__ newsfeat) {
  __shared__ int      s_nid[16];
  __shared__ int      s_title[16][TITLE_L];
  __shared__ _Float16 s_emb[16][TITLE_L][52];           // padded 50->52
  __shared__ float    s_cw[8][2][20];
  __shared__ float    s_cb[8];
  __shared__ __align__(16) _Float16 s_A[16 * FLATK];    // conv features
  __shared__ __align__(16) _Float16 s_A2[16 * 128];     // dense output

  const int t = threadIdx.x;
  const int rowbase = blockIdx.x * 16;

  if (t < 16) {
    int r = rowbase + t;
    int nid;
    if (r < BATCH) nid = news_indices[r];
    else { int q = r - BATCH; nid = user_news[user_indices[q / NB] * NB + q % NB]; }
    s_nid[t] = nid;
  }
  if (t < 320) ((float*)s_cw)[t] = conv_w[t];
  if (t >= 320 && t < 328) s_cb[t - 320] = conv_b[t - 320];
  __syncthreads();

  for (int idx = t; idx < 16 * TITLE_L; idx += 256)
    s_title[idx / TITLE_L][idx % TITLE_L] =
        title[s_nid[idx / TITLE_L] * TITLE_L + idx % TITLE_L];
  __syncthreads();

  for (int idx = t; idx < 16 * TITLE_L * WEMB; idx += 256) {
    int row = idx / (TITLE_L * WEMB), rem = idx % (TITLE_L * WEMB);
    int w = rem / WEMB, d = rem % WEMB;
    s_emb[row][w][d] = (_Float16)word_emb[s_title[row][w] * WEMB + d];
  }
  __syncthreads();

  { // conv + relu + maxpool(2,1): k = ch*124 + ph*31 + pw
    const int row = t >> 4, s = t & 15;
    for (int k = s; k < FLATK; k += 16) {
      int ch = k / 124, rem = k % 124, ph = rem / 31, pw = rem % 31;
      float m = -3.4e38f;
      for (int h2 = 0; h2 < 2; ++h2) {
        int h = 2 * ph + h2;
        float acc = s_cb[ch];
        for (int i = 0; i < 2; ++i)
          for (int j = 0; j < 20; ++j)
            acc += s_cw[ch][i][j] * (float)s_emb[row][h + i][pw + j];
        m = fmaxf(m, acc);
      }
      s_A[row * FLATK + k] = (_Float16)fmaxf(m, 0.f);   // relu(max) == max(relu)
    }
  }
  __syncthreads();

  const int lane = t & 31, wave = t >> 5;
  const int laneM = lane & 15, base8 = (lane < 16) ? 0 : 8;
  const int nt = wave, n = nt * 16 + laneM;

  { // GEMM1: [16x992] x dense_w -> relu -> s_A2 (fp16)
    v8f acc = bcast8(dense_b[n]);
    for (int kt = 0; kt < 31; ++kt) {
      v16h a = load_a_frag(s_A, laneM, base8, kt, FLATK);
      v16h b = *(const v16h*)(dsw + ((kt * 8 + nt) * 32 + lane) * 16);
      acc = wmma16(a, b, acc);
    }
    for (int r = 0; r < 8; ++r)
      s_A2[(r + base8) * 128 + n] = (_Float16)fmaxf(acc[r], 0.f);
  }
  __syncthreads();

  { // GEMM2: item transform + relu -> global fp16
    v8f acc = bcast8(item_b[n]);
    for (int kt = 0; kt < 4; ++kt) {
      v16h a = load_a_frag(s_A2, laneM, base8, kt, 128);
      v16h b = *(const v16h*)(isw + ((kt * 8 + nt) * 32 + lane) * 16);
      acc = wmma16(a, b, acc);
    }
    for (int r = 0; r < 8; ++r)
      newsfeat[(size_t)(rowbase + r + base8) * 128 + n] = (_Float16)fmaxf(acc[r], 0.f);
  }
}

// ---------------- K2: user-emb gather + user transform (128->128) -----------
__global__ __launch_bounds__(256) void user_encoder_kernel(
    const int* __restrict__ user_indices, const int* __restrict__ news_indices,
    const int* __restrict__ news_user,    const float* __restrict__ user_emb,
    const _Float16* __restrict__ usw,     const float* __restrict__ user_b,
    _Float16* __restrict__ userfeat) {
  __shared__ int s_uid[16];
  __shared__ __align__(16) _Float16 s_A[16 * 128];
  const int t = threadIdx.x;
  const int rowbase = blockIdx.x * 16;

  if (t < 16) {
    int r = rowbase + t;
    int uid;
    if (r < BATCH) uid = user_indices[r];
    else { int q = r - BATCH; uid = news_user[news_indices[q / NB] * NB + q % NB]; }
    s_uid[t] = uid;
  }
  __syncthreads();
  for (int idx = t; idx < 16 * 128; idx += 256)
    s_A[idx] = (_Float16)user_emb[(size_t)s_uid[idx >> 7] * 128 + (idx & 127)];
  __syncthreads();

  const int lane = t & 31, wave = t >> 5;
  const int laneM = lane & 15, base8 = (lane < 16) ? 0 : 8;
  const int nt = wave, n = nt * 16 + laneM;

  v8f acc = bcast8(user_b[n]);
  for (int kt = 0; kt < 4; ++kt) {
    v16h a = load_a_frag(s_A, laneM, base8, kt, 128);
    v16h b = *(const v16h*)(usw + ((kt * 8 + nt) * 32 + lane) * 16);
    acc = wmma16(a, b, acc);
  }
  for (int r = 0; r < 8; ++r)
    userfeat[(size_t)(rowbase + r + base8) * 128 + n] = (_Float16)fmaxf(acc[r], 0.f);
}

// ---------------- K3: capsule routing (4 iters, 30 neighbors, 8x16 caps) ----
__global__ __launch_bounds__(128) void routing_kernel(
    const _Float16* __restrict__ newsfeat, const _Float16* __restrict__ userfeat,
    _Float16* __restrict__ routed) {
  const int mode = blockIdx.x >> 11;    // 0: news routing, 1: user routing
  const int b    = blockIdx.x & 2047;
  const _Float16* xf = mode ? userfeat : newsfeat;  // self vector
  const _Float16* zf = mode ? newsfeat : userfeat;  // neighbor vectors
  const int t = threadIdx.x, c = t >> 4, d16 = t & 15;
  __shared__ float s_p[NB][NCAPS];

  float x = (float)xf[(size_t)b * 128 + t];
  float ss = x * x;
  for (int m = 8; m >= 1; m >>= 1) ss += __shfl_xor(ss, m, 16);
  x = x / (sqrtf(ss) + 1e-9f);

  float z[NB];
  for (int k = 0; k < NB; ++k) {
    float v = (float)zf[(size_t)(BATCH + b * NB + k) * 128 + t];
    float s2 = v * v;
    for (int m = 8; m >= 1; m >>= 1) s2 += __shfl_xor(s2, m, 16);
    z[k] = v / (sqrtf(s2) + 1e-9f);
  }

  float u = x;
  for (int it = 0; it < ROUTIT; ++it) {
    for (int k = 0; k < NB; ++k) {                 // logits <z_kc, u_c>
      float p = z[k] * u;
      for (int m = 8; m >= 1; m >>= 1) p += __shfl_xor(p, m, 16);
      if (d16 == 0) s_p[k][c] = p;
    }
    __syncthreads();
    if (t < NB) {                                  // softmax over caps
      float mx = -3.4e38f;
      for (int cc = 0; cc < NCAPS; ++cc) mx = fmaxf(mx, s_p[t][cc]);
      float sum = 0.f, e[NCAPS];
      for (int cc = 0; cc < NCAPS; ++cc) { e[cc] = expf(s_p[t][cc] - mx); sum += e[cc]; }
      for (int cc = 0; cc < NCAPS; ++cc) s_p[t][cc] = e[cc] / sum;
    }
    __syncthreads();
    float acc = x;
    for (int k = 0; k < NB; ++k) acc += s_p[k][c] * z[k];
    float s2 = acc * acc;
    for (int m = 8; m >= 1; m >>= 1) s2 += __shfl_xor(s2, m, 16);
    u = acc / (sqrtf(s2) + 1e-9f);
    __syncthreads();
  }
  routed[(size_t)mode * BATCH * 128 + (size_t)b * 128 + t] = (_Float16)u;
}

// ---------------- K4: last_w maps + row dot + sigmoid -----------------------
__global__ __launch_bounds__(256) void score_kernel(
    const _Float16* __restrict__ routed, const _Float16* __restrict__ lsw,
    const float* __restrict__ last_b, float* __restrict__ out) {
  __shared__ __align__(16) _Float16 sx[16 * 128], sy[16 * 128];
  __shared__ float rX[16 * 128], rY[16 * 128];
  const int t = threadIdx.x;
  const int b0 = blockIdx.x * 16;

  for (int idx = t; idx < 16 * 128; idx += 256) {
    int row = idx >> 7, col = idx & 127;
    sx[idx] = routed[(size_t)BATCH * 128 + (size_t)(b0 + row) * 128 + col]; // user_emb
    sy[idx] = routed[(size_t)(b0 + row) * 128 + col];                      // news_emb
  }
  __syncthreads();

  const int lane = t & 31, wave = t >> 5;
  const int laneM = lane & 15, base8 = (lane < 16) ? 0 : 8;
  const int nt = wave, n = nt * 16 + laneM;

  float bv = last_b[n];
  v8f aX = bcast8(bv), aY = bcast8(bv);
  for (int kt = 0; kt < 4; ++kt) {
    v16h b = *(const v16h*)(lsw + ((kt * 8 + nt) * 32 + lane) * 16);
    aX = wmma16(load_a_frag(sx, laneM, base8, kt, 128), b, aX);
    aY = wmma16(load_a_frag(sy, laneM, base8, kt, 128), b, aY);
  }
  for (int r = 0; r < 8; ++r) {
    rX[(r + base8) * 128 + n] = aX[r];
    rY[(r + base8) * 128 + n] = aY[r];
  }
  __syncthreads();

  const int row = t >> 4, l16 = t & 15;
  float s = 0.f;
  for (int j = l16 * 8; j < l16 * 8 + 8; ++j) s += rX[row * 128 + j] * rY[row * 128 + j];
  for (int m = 8; m >= 1; m >>= 1) s += __shfl_xor(s, m, 16);
  if (l16 == 0) out[b0 + row] = 1.f / (1.f + expf(-s));
}

// ---------------- workspace layout (fp16 element offsets) -------------------
#define OFF_DSW   ((size_t)0)                        // 31*8*512 = 126976
#define OFF_USW   (OFF_DSW + (size_t)31 * 8 * 512)   // 4*8*512  = 16384
#define OFF_ISW   (OFF_USW + (size_t)4 * 8 * 512)
#define OFF_LSW   (OFF_ISW + (size_t)4 * 8 * 512)
#define OFF_NEWS  (OFF_LSW + (size_t)4 * 8 * 512)    // ROWS*128
#define OFF_USER  (OFF_NEWS + (size_t)ROWS * 128)
#define OFF_ROUT  (OFF_USER + (size_t)ROWS * 128)    // 2*BATCH*128; end ~33.9 MB

extern "C" void kernel_launch(void* const* d_in, const int* in_sizes, int n_in,
                              void* d_out, int out_size, void* d_ws, size_t ws_size,
                              hipStream_t stream) {
  const int*   user_indices = (const int*)d_in[0];
  const int*   news_indices = (const int*)d_in[1];
  // d_in[2] = labels (unused by forward)
  const int*   news_user    = (const int*)d_in[3];
  const int*   user_news    = (const int*)d_in[4];
  const int*   title        = (const int*)d_in[5];
  const float* user_emb     = (const float*)d_in[6];
  const float* word_emb     = (const float*)d_in[7];
  const float* conv_w       = (const float*)d_in[8];
  const float* conv_b       = (const float*)d_in[9];
  const float* dense_w      = (const float*)d_in[10];
  const float* dense_b      = (const float*)d_in[11];
  const float* user_w       = (const float*)d_in[12];
  const float* user_b       = (const float*)d_in[13];
  const float* item_w       = (const float*)d_in[14];
  const float* item_b       = (const float*)d_in[15];
  const float* last_w       = (const float*)d_in[16];
  const float* last_b       = (const float*)d_in[17];
  float* out = (float*)d_out;

  _Float16* ws = (_Float16*)d_ws;
  _Float16* dsw      = ws + OFF_DSW;
  _Float16* usw      = ws + OFF_USW;
  _Float16* isw      = ws + OFF_ISW;
  _Float16* lsw      = ws + OFF_LSW;
  _Float16* newsfeat = ws + OFF_NEWS;
  _Float16* userfeat = ws + OFF_USER;
  _Float16* routed   = ws + OFF_ROUT;

  // K0: pack all weights into fp16 WMMA B-fragment order
  {
    int tot = 31 * 8 * 512;
    swizzle_w_kernel<<<(tot + 255) / 256, 256, 0, stream>>>(dense_w, dsw, 8, 128, tot);
    tot = 4 * 8 * 512;
    swizzle_w_kernel<<<(tot + 255) / 256, 256, 0, stream>>>(user_w, usw, 8, 128, tot);
    swizzle_w_kernel<<<(tot + 255) / 256, 256, 0, stream>>>(item_w, isw, 8, 128, tot);
    swizzle_w_kernel<<<(tot + 255) / 256, 256, 0, stream>>>(last_w, lsw, 8, 128, tot);
  }

  // K1 / K2: both sides' encoders (ROWS = 63488 rows, 16 rows per workgroup)
  news_encoder_kernel<<<ROWS / 16, 256, 0, stream>>>(
      news_indices, user_indices, user_news, title, word_emb,
      conv_w, conv_b, dsw, dense_b, isw, item_b, newsfeat);
  user_encoder_kernel<<<ROWS / 16, 256, 0, stream>>>(
      user_indices, news_indices, news_user, user_emb, usw, user_b, userfeat);

  // K3: capsule routing, 2 * BATCH instances
  routing_kernel<<<2 * BATCH, 128, 0, stream>>>(newsfeat, userfeat, routed);

  // K4: scoring
  score_kernel<<<BATCH / 16, 256, 0, stream>>>(routed, lsw, last_b, out);
}
